// CondConv_51032801411528
// MI455X (gfx1250) — matrix operand order repaired and checked by reading, not compiled, not run on hardware
//
#include <hip/hip_runtime.h>

// ---------------------------------------------------------------------------
// CondConv on MI455X (gfx1250), wave32, WMMA bf16 16x16x32.
//
// Pipeline (all on `stream`):
//   1) pool_convert : x(f32) -> x_bf16, plus global avg-pool -> pooled (B,I)
//   2) routing      : pooled -> MLP -> softmax r (B,E); cb = r @ bias (B,O)
//   3) combine      : cw_bf16[b] = sum_e r[b,e] * weight[e]   (18.9 MB bf16)
//   4) condconv     : per-sample implicit GEMM conv, M=256,N=3136,K=1152,
//                     256x64 block tile, K-step 32, double-buffered LDS,
//                     8x v_wmma_f32_16x16x32_bf16 per wave per K-step
// ---------------------------------------------------------------------------

typedef __bf16 bf16_t;
typedef __attribute__((ext_vector_type(16))) __bf16 v16bf;
typedef __attribute__((ext_vector_type(8)))  __bf16 v8bf;
typedef __attribute__((ext_vector_type(8)))  float  v8f;

#define NB   32          // batch
#define NI   128         // in channels
#define NO   256         // out channels
#define NE   8           // experts
#define NHW  3136        // 56*56
#define NK9  1152        // NI*9
#define KSTEPS 36        // NK9/32
#define WELM 294912      // NO*NI*9

__device__ __forceinline__ bf16_t f2bf(float f) {
  union { float f; unsigned u; } c; c.f = f;
  unsigned r = (c.u + 0x7FFFu + ((c.u >> 16) & 1u)) >> 16;   // RNE
  union { unsigned short s; bf16_t b; } o; o.s = (unsigned short)r;
  return o.b;
}

// ---- 1) fused f32->bf16 convert + global average pool ---------------------
__global__ __launch_bounds__(256) void pool_convert(
    const float* __restrict__ x, bf16_t* __restrict__ xb,
    float* __restrict__ pooled) {
  const int blk = blockIdx.x;                       // b*NI + i
  const float* src = x + (size_t)blk * NHW;
  bf16_t* dst = xb + (size_t)blk * NHW;
  float s = 0.f;
  for (int p = threadIdx.x; p < NHW; p += 256) {
    float v = src[p];
    dst[p] = f2bf(v);
    s += v;
  }
  __shared__ float sm[256];
  sm[threadIdx.x] = s;
  __syncthreads();
  for (int off = 128; off > 0; off >>= 1) {
    if (threadIdx.x < off) sm[threadIdx.x] += sm[threadIdx.x + off];
    __syncthreads();
  }
  if (threadIdx.x == 0) pooled[blk] = sm[0] * (1.0f / (float)NHW);
}

// ---- 2) routing MLP + softmax + combined bias -----------------------------
__global__ __launch_bounds__(256) void routing_kernel(
    const float* __restrict__ pooled, const float* __restrict__ w1,
    const float* __restrict__ b1, const float* __restrict__ w2,
    const float* __restrict__ b2, const float* __restrict__ bias,
    float* __restrict__ r_out, float* __restrict__ cb_out) {
  __shared__ float rs[NB][NE];
  const int t = threadIdx.x;
  if (t < NB) {
    const int b = t;
    float h[32];
    const float* pr = pooled + b * NI;
    for (int j = 0; j < 32; ++j) {
      float s = b1[j];
      const float* wr = w1 + j * NI;
      for (int i = 0; i < NI; ++i) s += pr[i] * wr[i];
      h[j] = s > 0.f ? s : 0.f;
    }
    float lg[NE]; float mx = -1e30f;
    for (int e = 0; e < NE; ++e) {
      float s = b2[e];
      const float* wr = w2 + e * 32;
      for (int j = 0; j < 32; ++j) s += h[j] * wr[j];
      lg[e] = s; mx = fmaxf(mx, s);
    }
    float sum = 0.f;
    for (int e = 0; e < NE; ++e) { lg[e] = __expf(lg[e] - mx); sum += lg[e]; }
    const float inv = 1.0f / sum;
    for (int e = 0; e < NE; ++e) {
      float v = lg[e] * inv;
      rs[b][e] = v;
      r_out[b * NE + e] = v;
    }
  }
  __syncthreads();
  for (int idx = t; idx < NB * NO; idx += 256) {
    int b = idx >> 8, o = idx & (NO - 1);
    float s = 0.f;
#pragma unroll
    for (int e = 0; e < NE; ++e) s += rs[b][e] * bias[e * NO + o];
    cb_out[idx] = s;
  }
}

// ---- 3) per-sample combined weights, stored bf16 --------------------------
__global__ __launch_bounds__(256) void combine_weights(
    const float* __restrict__ r, const float* __restrict__ weight,
    bf16_t* __restrict__ cwb) {
  const size_t idx = (size_t)blockIdx.x * 256 + threadIdx.x;  // < NB*WELM
  const int b   = (int)(idx / WELM);
  const int rem = (int)(idx - (size_t)b * WELM);
  const float* rb = r + b * NE;
  float s = 0.f;
#pragma unroll
  for (int e = 0; e < NE; ++e) s += rb[e] * weight[(size_t)e * WELM + rem];
  cwb[idx] = f2bf(s);
}

// ---- 4) implicit-GEMM conv with bf16 WMMA, double-buffered LDS ------------
// grid = (49, 32): (N tiles of 64 pixels, batch). Block tile: 256(M) x 64(N).
// 8 waves; wave tile 64(M) x 32(N) => 8 WMMAs / wave / K-step.
__global__ __launch_bounds__(256) void condconv_wmma(
    const bf16_t* __restrict__ xb, const bf16_t* __restrict__ cwb,
    const float* __restrict__ cb, float* __restrict__ out) {
  const int b  = blockIdx.y;
  const int p0 = blockIdx.x * 64;

  __shared__ __align__(16) bf16_t As[2][256][40];   // [buf][m][k] pad 32->40
  __shared__ __align__(16) bf16_t Bs[2][64][40];    // [buf][n][k] N-major

  const int tid  = threadIdx.x;
  const int wave = tid >> 5, lane = tid & 31;
  // wave -> 4 M strips x 2 N tiles (64 output tiles / 8 waves = 8 WMMAs each)
  const int mq  = wave >> 1;               // M quad: strips 4*mq .. 4*mq+3
  const int ni0 = (wave & 1) * 2, ni1 = ni0 + 1;

  v8f acc[4][2] = {};

  const bf16_t* cwB = cwb + (size_t)b * WELM;       // all 256 out channels
  const bf16_t* xB  = xb + (size_t)b * NI * NHW;

  // A fill: 256 rows x 32 k; each thread owns one full row (64B contiguous)
  const int am = tid;                      // 0..255
  const bf16_t* aSrcBase = cwB + (size_t)am * NK9;

  // B fill: one k per thread-octet, 8 consecutive pixels per thread
  const int kk  = tid >> 3;                // 0..31
  const int pq  = (tid & 7) * 8;           // 0..56 step 8
  const int pb  = p0 + pq;
  const int ph0 = pb / 56, pw0 = pb - (pb / 56) * 56;

  // fragment addressing per ISA 7.12.2 (wave32 bf16 layouts)
  const int lan  = lane & 15;
  const int a_kb = (lane < 16) ? 0 : 8;    // A: K[0..7]+[16..23] / K[8..15]+[24..31]
  const int b_kb = (lane < 16) ? 0 : 8;    // B: K[0..7]+[16..23] / K[8..15]+[24..31], N = lane%16
  const int ma_base = mq * 64 + lan;       // + 16*j for strip j
  const int n0  = ni0 * 16 + lan, n1 = ni1 * 16 + lan;
  const bf16_t bz = f2bf(0.0f);

  v8bf areg[4];
  bf16_t breg[8];

  // staged loads for K-step k0 (into registers, division-light im2col)
  auto load_step = [&](int k0) {
    const v8bf* ap = (const v8bf*)(aSrcBase + k0);
#pragma unroll
    for (int q = 0; q < 4; ++q) areg[q] = ap[q];
    __builtin_prefetch(aSrcBase + k0 + 32, 0, 1);  // next K-step A stream
    int k  = k0 + kk;
    int i  = k / 9;
    int r9 = k - i * 9;
    int dh = (r9 >= 6) ? 1 : ((r9 >= 3) ? 0 : -1);
    int dw = r9 - (dh + 1) * 3 - 1;
    const bf16_t* xrow = xB + (size_t)i * NHW;
    int h = ph0, w = pw0;
#pragma unroll
    for (int q = 0; q < 8; ++q) {
      int hh = h + dh, ww = w + dw;
      bf16_t v = bz;
      if ((unsigned)hh < 56u && (unsigned)ww < 56u) v = xrow[hh * 56 + ww];
      breg[q] = v;
      if (++w == 56) { w = 0; ++h; }
    }
  };
  auto store_step = [&](int buf) {
    v8bf* adst = (v8bf*)&As[buf][am][0];
#pragma unroll
    for (int q = 0; q < 4; ++q) adst[q] = areg[q];
#pragma unroll
    for (int q = 0; q < 8; ++q) Bs[buf][pq + q][kk] = breg[q];
  };

  load_step(0);
  store_step(0);
  __syncthreads();

  for (int s = 0; s < KSTEPS; ++s) {
    const int cur = s & 1;
    const bool more = (s + 1) < KSTEPS;
    if (more) load_step((s + 1) * 32);             // global loads issue early

    union F { v16bf v; v8bf h[2]; };
    F fa[4], fb[2];
#pragma unroll
    for (int j = 0; j < 4; ++j) {
      const int ma = ma_base + j * 16;
      fa[j].h[0] = *(const v8bf*)&As[cur][ma][a_kb];
      fa[j].h[1] = *(const v8bf*)&As[cur][ma][a_kb + 16];
    }
    fb[0].h[0] = *(const v8bf*)&Bs[cur][n0][b_kb];
    fb[0].h[1] = *(const v8bf*)&Bs[cur][n0][b_kb + 16];
    fb[1].h[0] = *(const v8bf*)&Bs[cur][n1][b_kb];
    fb[1].h[1] = *(const v8bf*)&Bs[cur][n1][b_kb + 16];

#pragma unroll
    for (int j = 0; j < 4; ++j) {
#pragma unroll
      for (int t = 0; t < 2; ++t) {
        acc[j][t] = __builtin_amdgcn_wmma_f32_16x16x32_bf16(
            false, fa[j].v, false, fb[t].v, (short)0, acc[j][t], false, false);
      }
    }

    if (more) store_step(cur ^ 1);
    __syncthreads();
  }

  // epilogue: C/D layout (VGPR r -> M=r+8*(lane>=16), N=lane%16) + bias
  const float* cbb = cb + b * NO;
  const int mofs = (lane >> 4) << 3;
#pragma unroll
  for (int j = 0; j < 4; ++j) {
#pragma unroll
    for (int rr = 0; rr < 8; ++rr) {
      int o = (mq * 4 + j) * 16 + rr + mofs;       // o_base = 0 (full M)
      float bv = cbb[o];
      size_t base = ((size_t)(b * NO + o)) * NHW + p0;
      out[base + ni0 * 16 + lan] = acc[j][0][rr] + bv;
      out[base + ni1 * 16 + lan] = acc[j][1][rr] + bv;
    }
  }
}

// ---------------------------------------------------------------------------
extern "C" void kernel_launch(void* const* d_in, const int* in_sizes, int n_in,
                              void* d_out, int out_size, void* d_ws, size_t ws_size,
                              hipStream_t stream) {
  (void)in_sizes; (void)n_in; (void)out_size; (void)ws_size;
  const float* x      = (const float*)d_in[0];   // (32,128,56,56)
  const float* weight = (const float*)d_in[1];   // (8,256,128,3,3)
  const float* bias   = (const float*)d_in[2];   // (8,256)
  const float* w1     = (const float*)d_in[3];   // (32,128)
  const float* b1     = (const float*)d_in[4];   // (32,)
  const float* w2     = (const float*)d_in[5];   // (8,32)
  const float* b2     = (const float*)d_in[6];   // (8,)
  float* out = (float*)d_out;                    // (32,256,56,56)

  // workspace layout (≈44.6 MB, 256B-aligned segments)
  char* ws = (char*)d_ws;
  bf16_t* xb     = (bf16_t*)(ws);                       // 25,690,112 B
  bf16_t* cwb    = (bf16_t*)(ws + 25690112);            // 18,874,368 B
  float*  pooled = (float*) (ws + 44564480);            // 16,384 B
  float*  rbuf   = (float*) (ws + 44580864);            // 1,024 B
  float*  cbuf   = (float*) (ws + 44581888);            // 32,768 B

  pool_convert   <<<NB * NI, 256, 0, stream>>>(x, xb, pooled);
  routing_kernel <<<1,       256, 0, stream>>>(pooled, w1, b1, w2, b2, bias,
                                               rbuf, cbuf);
  combine_weights<<<(NB * WELM) / 256, 256, 0, stream>>>(rbuf, weight, cwb);

  dim3 grid(NHW / 64, NB);   // (49, 32)
  condconv_wmma  <<<grid,    256, 0, stream>>>(xb, cwb, cbuf, out);
}
